// SparseMoE_58463094833556
// MI455X (gfx1250) — matrix operand order repaired and checked by reading, not compiled, not run on hardware
//
#include <hip/hip_runtime.h>
#include <hip/hip_bf16.h>

// ---------------- problem constants (match reference) ----------------
#define DMODEL   1024
#define NEXP     8
#define TOPK     2
#define HIDDEN   4096
#define NTOK     8192          // B*T = 4*2048
#define CAPACITY 2048          // ceil(2.0 * 8192 / 8)

// ---------------- vector types for WMMA ----------------
typedef __attribute__((ext_vector_type(16))) __bf16          v16bf;
typedef __attribute__((ext_vector_type(8)))  float           v8f;
typedef __attribute__((ext_vector_type(8)))  unsigned short  u16x8;
typedef __attribute__((ext_vector_type(16))) unsigned short  u16x16;

// fp32 -> bf16 round-to-nearest-even
__device__ __forceinline__ unsigned short f2bf(float f) {
  unsigned int u = __float_as_uint(f);
  u += 0x7fffu + ((u >> 16) & 1u);
  return (unsigned short)(u >> 16);
}

// CDNA5 async global->LDS copy: each active lane moves 16 bytes from its
// global address to its LDS offset. Tracked by ASYNCcnt.
__device__ __forceinline__ void async_copy_b128(unsigned ldsOff, const void* gaddr) {
  asm volatile("global_load_async_to_lds_b128 %0, %1, off"
               :
               : "v"(ldsOff), "v"(gaddr)
               : "memory");
}
__device__ __forceinline__ void wait_async0() {
  asm volatile("s_wait_asynccnt 0" ::: "memory");
}

// =====================================================================
// 1) Transpose + fp32->bf16 convert:  in (E,R,C) f32  ->  out (E,C,R) bf16
// =====================================================================
__global__ __launch_bounds__(256) void transpose_cvt_kernel(
    const float* __restrict__ in, unsigned short* __restrict__ out, int R, int C) {
  __shared__ float tile[32][33];
  size_t base = (size_t)blockIdx.z * (size_t)R * (size_t)C;
  int c0 = blockIdx.x * 32, r0 = blockIdx.y * 32;
  int tx = threadIdx.x, ty = threadIdx.y;
#pragma unroll
  for (int j = 0; j < 32; j += 8)
    tile[ty + j][tx] = in[base + (size_t)(r0 + ty + j) * C + (c0 + tx)];
  __syncthreads();
#pragma unroll
  for (int j = 0; j < 32; j += 8)
    out[base + (size_t)(c0 + ty + j) * R + (r0 + tx)] = f2bf(tile[tx][ty + j]);
}

// =====================================================================
// 2) Router: one wave32 per token. logits = x@gate_w, softmax, top-2.
// =====================================================================
__global__ __launch_bounds__(256) void router_kernel(
    const float* __restrict__ x, const float* __restrict__ gw,
    float* __restrict__ probs, int* __restrict__ eids) {
  int tok  = (blockIdx.x * blockDim.x + threadIdx.x) >> 5;
  int lane = threadIdx.x & 31;
  if (tok >= NTOK) return;
  const float* xr = x + (size_t)tok * DMODEL;
  float acc[NEXP];
#pragma unroll
  for (int e = 0; e < NEXP; ++e) acc[e] = 0.f;
  for (int d = lane; d < DMODEL; d += 32) {
    float xv = xr[d];
#pragma unroll
    for (int e = 0; e < NEXP; ++e) acc[e] += xv * gw[d * NEXP + e];
  }
#pragma unroll
  for (int off = 16; off > 0; off >>= 1) {
#pragma unroll
    for (int e = 0; e < NEXP; ++e) acc[e] += __shfl_down(acc[e], off, 32);
  }
  if (lane == 0) {
    float mx = acc[0];
#pragma unroll
    for (int e = 1; e < NEXP; ++e) mx = fmaxf(mx, acc[e]);
    float ex[NEXP]; float s = 0.f;
#pragma unroll
    for (int e = 0; e < NEXP; ++e) { ex[e] = expf(acc[e] - mx); s += ex[e]; }
    float inv = 1.f / s;
    int i1 = 0; float v1 = -1.f;
#pragma unroll
    for (int e = 0; e < NEXP; ++e) { float g = ex[e] * inv; if (g > v1) { v1 = g; i1 = e; } }
    int i2 = (i1 == 0) ? 1 : 0; float v2 = -1.f;
#pragma unroll
    for (int e = 0; e < NEXP; ++e) {
      if (e == i1) continue;
      float g = ex[e] * inv; if (g > v2) { v2 = g; i2 = e; }
    }
    eids[tok * 2 + 0] = i1; eids[tok * 2 + 1] = i2;
    probs[tok * 2 + 0] = v1; probs[tok * 2 + 1] = v2;
  }
}

// =====================================================================
// 3) Deterministic capacity assignment: one block per expert, ordered
//    Hillis-Steele scan over tokens (matches reference cumsum semantics).
// =====================================================================
__global__ __launch_bounds__(256) void assign_kernel(
    const int* __restrict__ eids, int* __restrict__ slots) {
  int e = blockIdx.x;
  __shared__ int sdata[256];
  __shared__ int running;
  if (threadIdx.x == 0) running = 0;
  __syncthreads();
  for (int base = 0; base < NTOK; base += 256) {
    int n  = base + threadIdx.x;
    int kk = -1;
    if      (eids[n * 2 + 0] == e) kk = 0;
    else if (eids[n * 2 + 1] == e) kk = 1;
    int ind = (kk >= 0) ? 1 : 0;
    sdata[threadIdx.x] = ind;
    __syncthreads();
#pragma unroll
    for (int off = 1; off < 256; off <<= 1) {
      int v = (threadIdx.x >= off) ? sdata[threadIdx.x - off] : 0;
      __syncthreads();
      sdata[threadIdx.x] += v;
      __syncthreads();
    }
    int pos = running + sdata[threadIdx.x] - 1;   // inclusive -> position
    if (kk >= 0) slots[n * 2 + kk] = (pos < CAPACITY) ? pos : -1;
    __syncthreads();
    if (threadIdx.x == 0) running += sdata[255];
    __syncthreads();
  }
}

// =====================================================================
// 4) Dispatch kept tokens into bf16 capacity buffers (E, CAP, D)
// =====================================================================
__global__ __launch_bounds__(256) void dispatch_kernel(
    const float* __restrict__ x, const int* __restrict__ eids,
    const int* __restrict__ slots, unsigned short* __restrict__ disp) {
  int a    = (blockIdx.x * blockDim.x + threadIdx.x) >> 5;   // (token,k) pair
  int lane = threadIdx.x & 31;
  int s = slots[a];
  if (s < 0) return;
  int e = eids[a];
  int n = a >> 1;
  const float* src = x + (size_t)n * DMODEL;
  unsigned short* dst = disp + ((size_t)e * CAPACITY + s) * DMODEL;
  for (int d = lane * 4; d < DMODEL; d += 128) {
    float4 v = *reinterpret_cast<const float4*>(src + d);
    dst[d + 0] = f2bf(v.x); dst[d + 1] = f2bf(v.y);
    dst[d + 2] = f2bf(v.z); dst[d + 3] = f2bf(v.w);
  }
}

// =====================================================================
// 5) Grouped GEMM with v_wmma_f32_16x16x32_bf16, async global->LDS
//    double-buffered staging.
//    A: (E, M, K) bf16 row-major.  B: (E, N, K) bf16 (pre-transposed, K
//    contiguous).  Block = 256 threads (8 waves) -> 128x64 output tile;
//    each wave owns 16 rows x 64 cols (4 accumulators).
//    Per k-step (32): stage A 128x32 (8KB) + B 64x32 (4KB) into LDS via
//    GLOBAL_LOAD_ASYNC_TO_LDS_B128 (coalesced 16B/lane), overlap next
//    tile's DMA with current tile's WMMAs. All fragment ds_loads are
//    hoisted ahead of the WMMA chain so the 4 WMMAs issue back-to-back
//    behind a single DScnt wait instead of 4 serializing waits.
//    Fragment layouts per CDNA5 ISA 7.12.2 (h = lane>>4, l15 = lane&15):
//      A lane: row m0+l15; K chunks [h*8,h*8+8) and [h*8+16,h*8+24)
//      B lane: col n0+l15; K chunks [h*8,h*8+8) and [h*8+16,h*8+24)
//      C vgpr v: row m0 + v + h*8, col n0 + l15
// =====================================================================
template <bool GELU_BF16_OUT>
__global__ __launch_bounds__(256) void moe_gemm_kernel(
    const unsigned short* __restrict__ Aall,
    const unsigned short* __restrict__ Ball,
    const float* __restrict__ biasAll,
    void* __restrict__ OutAll,
    int M, int Nn, int Kk) {
  __shared__ unsigned short sA[2][128 * 32];   // 2 x 8KB
  __shared__ unsigned short sB[2][64 * 32];    // 2 x 4KB

  int e = blockIdx.z;
  const unsigned short* A  = Aall + (size_t)e * M  * Kk;
  const unsigned short* Bm = Ball + (size_t)e * Nn * Kk;
  const float* bias = biasAll + (size_t)e * Nn;

  int tid  = threadIdx.x;
  int wave = tid >> 5;
  int lane = tid & 31;
  int half = lane >> 4;
  int l15  = lane & 15;

  int mBase = blockIdx.y * 128;
  int n0    = blockIdx.x * 64;

  const unsigned short* Ablk = A  + (size_t)mBase * Kk;
  const unsigned short* Bblk = Bm + (size_t)n0 * Kk;

  // LDS byte offsets of the staging buffers (low 32 bits of flat address)
  unsigned sAoff = (unsigned)(unsigned long long)&sA[0][0];
  unsigned sBoff = (unsigned)(unsigned long long)&sB[0][0];

  // per-thread copy-chunk coordinates (16B chunks)
  int arow0 = tid >> 2, apart0 = tid & 3;            // chunks 0..255
  int arow1 = (tid + 256) >> 2, apart1 = tid & 3;    // chunks 256..511
  int bcol  = tid >> 2, bpart = tid & 3;

  v8f zero = {0.f, 0.f, 0.f, 0.f, 0.f, 0.f, 0.f, 0.f};
  v8f acc[4] = {zero, zero, zero, zero};

  // ---- prologue: stage k0 = 0 into buffer 0 ----
  async_copy_b128(sAoff + (unsigned)((arow0 * 32 + apart0 * 8) * 2),
                  Ablk + (size_t)arow0 * Kk + apart0 * 8);
  async_copy_b128(sAoff + (unsigned)((arow1 * 32 + apart1 * 8) * 2),
                  Ablk + (size_t)arow1 * Kk + apart1 * 8);
  async_copy_b128(sBoff + (unsigned)((bcol * 32 + bpart * 8) * 2),
                  Bblk + (size_t)bcol * Kk + bpart * 8);

  for (int k0 = 0; k0 < Kk; k0 += 32) {
    int p = (k0 >> 5) & 1;

    wait_async0();       // this wave's async writes complete
    __syncthreads();     // all waves' writes to buf p visible; prior reads of buf p^1 drained

    // stage next tile into the other buffer (overlaps with compute below)
    if (k0 + 32 < Kk) {
      int kn = k0 + 32;
      unsigned an = sAoff + (unsigned)((p ^ 1) * 128 * 32 * 2);
      unsigned bn = sBoff + (unsigned)((p ^ 1) * 64 * 32 * 2);
      async_copy_b128(an + (unsigned)((arow0 * 32 + apart0 * 8) * 2),
                      Ablk + (size_t)arow0 * Kk + kn + apart0 * 8);
      async_copy_b128(an + (unsigned)((arow1 * 32 + apart1 * 8) * 2),
                      Ablk + (size_t)arow1 * Kk + kn + apart1 * 8);
      async_copy_b128(bn + (unsigned)((bcol * 32 + bpart * 8) * 2),
                      Bblk + (size_t)bcol * Kk + kn + bpart * 8);
    }

    // ---- hoist ALL fragment loads (A + 4xB) ahead of the WMMA chain ----
    int arow = wave * 16 + l15;
    u16x8 alo = *reinterpret_cast<const u16x8*>(&sA[p][arow * 32 + half * 8]);
    u16x8 ahi = *reinterpret_cast<const u16x8*>(&sA[p][arow * 32 + half * 8 + 16]);
    union { u16x16 u; v16bf b; } fa;
    fa.u = __builtin_shufflevector(alo, ahi, 0, 1, 2, 3, 4, 5, 6, 7,
                                             8, 9, 10, 11, 12, 13, 14, 15);

    union { u16x16 u; v16bf b; } fb[4];
#pragma unroll
    for (int j = 0; j < 4; ++j) {
      int bcolj = j * 16 + l15;
      u16x8 blo = *reinterpret_cast<const u16x8*>(&sB[p][bcolj * 32 + half * 8]);
      u16x8 bhi = *reinterpret_cast<const u16x8*>(&sB[p][bcolj * 32 + half * 8 + 16]);
      fb[j].u = __builtin_shufflevector(blo, bhi, 0, 1, 2, 3, 4, 5, 6, 7,
                                                  8, 9, 10, 11, 12, 13, 14, 15);
    }

    acc[0] = __builtin_amdgcn_wmma_f32_16x16x32_bf16(false, fa.b, false, fb[0].b,
                                                     (short)0, acc[0], false, false);
    acc[1] = __builtin_amdgcn_wmma_f32_16x16x32_bf16(false, fa.b, false, fb[1].b,
                                                     (short)0, acc[1], false, false);
    acc[2] = __builtin_amdgcn_wmma_f32_16x16x32_bf16(false, fa.b, false, fb[2].b,
                                                     (short)0, acc[2], false, false);
    acc[3] = __builtin_amdgcn_wmma_f32_16x16x32_bf16(false, fa.b, false, fb[3].b,
                                                     (short)0, acc[3], false, false);
  }

  int m0 = mBase + wave * 16;
  int rowBase = m0 + half * 8;
  if constexpr (GELU_BF16_OUT) {
    unsigned short* O = reinterpret_cast<unsigned short*>(OutAll) +
                        (size_t)e * M * Nn;
#pragma unroll
    for (int j = 0; j < 4; ++j) {
      int col = n0 + j * 16 + l15;
      float bb = bias[col];
#pragma unroll
      for (int v = 0; v < 8; ++v) {
        float xv = acc[j][v] + bb;
        float g  = 0.5f * xv * (1.0f + erff(xv * 0.7071067811865475f)); // exact gelu
        O[(size_t)(rowBase + v) * Nn + col] = f2bf(g);
      }
    }
  } else {
    float* O = reinterpret_cast<float*>(OutAll) + (size_t)e * M * Nn;
#pragma unroll
    for (int j = 0; j < 4; ++j) {
      int col = n0 + j * 16 + l15;
      float bb = bias[col];
#pragma unroll
      for (int v = 0; v < 8; ++v)
        O[(size_t)(rowBase + v) * Nn + col] = acc[j][v] + bb;
    }
  }
}

// =====================================================================
// 6) Combine: out[n] = p0*out_e[e0][s0] + p1*out_e[e1][s1]  (0 if dropped)
// =====================================================================
__global__ __launch_bounds__(256) void combine_kernel(
    const float* __restrict__ oute, const float* __restrict__ probs,
    const int* __restrict__ eids, const int* __restrict__ slots,
    float* __restrict__ out) {
  if (blockIdx.x == 0 && threadIdx.x == 0)
    out[(size_t)NTOK * DMODEL] = 0.f;             // aux_loss = 0 (eval mode)
  int n    = (blockIdx.x * blockDim.x + threadIdx.x) >> 5;
  int lane = threadIdx.x & 31;
  if (n >= NTOK) return;
  int s0 = slots[2 * n + 0], s1 = slots[2 * n + 1];
  float p0 = (s0 >= 0) ? probs[2 * n + 0] : 0.f;
  float p1 = (s1 >= 0) ? probs[2 * n + 1] : 0.f;
  const float* r0 = oute + ((size_t)eids[2 * n + 0] * CAPACITY + (s0 < 0 ? 0 : s0)) * DMODEL;
  const float* r1 = oute + ((size_t)eids[2 * n + 1] * CAPACITY + (s1 < 0 ? 0 : s1)) * DMODEL;
  float* o = out + (size_t)n * DMODEL;
  for (int d = lane * 4; d < DMODEL; d += 128) {
    float4 a = *reinterpret_cast<const float4*>(r0 + d);
    float4 b = *reinterpret_cast<const float4*>(r1 + d);
    float4 r;
    r.x = p0 * a.x + p1 * b.x;
    r.y = p0 * a.y + p1 * b.y;
    r.z = p0 * a.z + p1 * b.z;
    r.w = p0 * a.w + p1 * b.w;
    *reinterpret_cast<float4*>(o + d) = r;
  }
}

// =====================================================================
// launcher
// =====================================================================
extern "C" void kernel_launch(void* const* d_in, const int* in_sizes, int n_in,
                              void* d_out, int out_size, void* d_ws, size_t ws_size,
                              hipStream_t stream) {
  (void)in_sizes; (void)n_in; (void)out_size; (void)ws_size;
  const float* x      = (const float*)d_in[0];   // (B,T,D)
  const float* gate_w = (const float*)d_in[1];   // (D,E)
  const float* w1     = (const float*)d_in[2];   // (E,D,HID)
  const float* b1     = (const float*)d_in[3];   // (E,HID)
  const float* w2     = (const float*)d_in[4];   // (E,HID,D)
  const float* b2     = (const float*)d_in[5];   // (E,D)
  float* out = (float*)d_out;                    // (B,T,D) floats + 1 aux scalar

  char* ws = (char*)d_ws;
  size_t off = 0;
  auto carve = [&](size_t bytes) -> char* {
    char* p = ws + off;
    off += (bytes + 255) & ~(size_t)255;
    return p;
  };
  float*          probs  = (float*)carve((size_t)NTOK * 2 * sizeof(float));
  int*            eids   = (int*)carve((size_t)NTOK * 2 * sizeof(int));
  int*            slots  = (int*)carve((size_t)NTOK * 2 * sizeof(int));
  unsigned short* w1t    = (unsigned short*)carve((size_t)NEXP * HIDDEN * DMODEL * 2);
  unsigned short* w2t    = (unsigned short*)carve((size_t)NEXP * DMODEL * HIDDEN * 2);
  unsigned short* dispBf = (unsigned short*)carve((size_t)NEXP * CAPACITY * DMODEL * 2);
  unsigned short* Hbuf   = (unsigned short*)carve((size_t)NEXP * CAPACITY * HIDDEN * 2);
  float*          oute   = (float*)carve((size_t)NEXP * CAPACITY * DMODEL * sizeof(float));

  // 1) weight transpose+convert: w1 (E,D,HID)->(E,HID,D), w2 (E,HID,D)->(E,D,HID)
  transpose_cvt_kernel<<<dim3(HIDDEN / 32, DMODEL / 32, NEXP), dim3(32, 8), 0, stream>>>(
      w1, w1t, DMODEL, HIDDEN);
  transpose_cvt_kernel<<<dim3(DMODEL / 32, HIDDEN / 32, NEXP), dim3(32, 8), 0, stream>>>(
      w2, w2t, HIDDEN, DMODEL);

  // 2) router (wave per token, 8 waves per block)
  router_kernel<<<NTOK / 8, 256, 0, stream>>>(x, gate_w, probs, eids);

  // 3) deterministic ordered capacity assignment
  assign_kernel<<<NEXP, 256, 0, stream>>>(eids, slots);

  // 4) dispatch to bf16 capacity buffers
  dispatch_kernel<<<(NTOK * TOPK) / 8, 256, 0, stream>>>(x, eids, slots, dispBf);

  // 5) grouped GEMM1 + bias + exact GeLU -> bf16 H
  moe_gemm_kernel<true><<<dim3(HIDDEN / 64, CAPACITY / 128, NEXP), 256, 0, stream>>>(
      dispBf, w1t, b1, Hbuf, CAPACITY, HIDDEN, DMODEL);

  // 6) grouped GEMM2 + bias -> f32 expert outputs
  moe_gemm_kernel<false><<<dim3(DMODEL / 64, CAPACITY / 128, NEXP), 256, 0, stream>>>(
      Hbuf, w2t, b2, oute, CAPACITY, DMODEL, HIDDEN);

  // 7) weighted combine + aux_loss
  combine_kernel<<<NTOK / 8, 256, 0, stream>>>(oute, probs, eids, slots, out);
}
